// MultiHeadedAttention_7791070675354
// MI455X (gfx1250) — compile-verified, matrix operand order checked by reading, not run on hardware
//
#include <hip/hip_runtime.h>
#include <hip/hip_fp16.h>

#define B_   2
#define S_   2048
#define D_   1024
#define H_   16
#define DK_  64
#define MTOT (B_ * S_)   // 4096

typedef __attribute__((ext_vector_type(16))) _Float16 v16h;
typedef __attribute__((ext_vector_type(8)))  _Float16 v8h;
typedef __attribute__((ext_vector_type(8)))  float    v8f;
typedef __attribute__((ext_vector_type(4)))  unsigned v4u;
typedef __attribute__((ext_vector_type(8)))  int      v8i;
typedef __attribute__((ext_vector_type(4)))  int      v4i;

// ---------------------------------------------------------------------------
// Tensor Data Mover: DMA a 2D f16 tile [tile_d1 rows x tile_d0 elems] from
// global (row stride stride0 elems) into LDS at byte offset lds_off,
// packed contiguously. Descriptor layout per CDNA5 ISA.
// Issued once per wave (EXEC ignored); tracked by TENSORcnt.
// ---------------------------------------------------------------------------
__device__ inline void tdm_load_2d(unsigned lds_off, const void* gptr,
                                   unsigned tile_d0, unsigned tile_d1,
                                   unsigned long long stride0,
                                   unsigned tensor_d0, unsigned tensor_d1) {
  unsigned long long ga = (unsigned long long)(uintptr_t)gptr;
  v4u g0;
  g0[0] = 1u;                                   // count=1 (valid descriptor)
  g0[1] = lds_off;                              // lds_addr (bytes)
  g0[2] = (unsigned)ga;                         // global_addr[31:0]
  g0[3] = (unsigned)((ga >> 32) & 0x01FFFFFFu)  // global_addr[56:32]
          | (2u << 30);                         // type = 2 ("image")
  v8i g1;
  g1[0] = (int)(1u << 16);                      // data_size=1 -> 2 bytes
  g1[1] = (int)((tensor_d0 & 0xFFFFu) << 16);   // tensor_dim0[15:0] @ [63:48]
  g1[2] = (int)((tensor_d0 >> 16) | ((tensor_d1 & 0xFFFFu) << 16));
  g1[3] = (int)((tensor_d1 >> 16) | (tile_d0 << 16));   // tile_dim0 @ [127:112]
  g1[4] = (int)tile_d1;                         // tile_dim1 (tile_dim2 = 0)
  g1[5] = (int)(unsigned)(stride0 & 0xFFFFFFFFu);       // dim0 stride lo
  g1[6] = (int)(unsigned)((stride0 >> 32) & 0xFFFFu);   // dim0 stride hi
  g1[7] = 0;
  v4i z4 = {};
#if __clang_major__ >= 23
  v8i z8 = {};
  __builtin_amdgcn_tensor_load_to_lds(g0, g1, z4, z4, z8, 0);
#else
  __builtin_amdgcn_tensor_load_to_lds(g0, g1, z4, z4, 0);
#endif
}

// ---------------------------------------------------------------------------
// Fragment loader: 16x32 f16 A-fragment (or symmetric B-fragment of a Kx16
// column-block with contiguous K). ISA layout (CDNA5 ISA): lane m = lane%16
// holds row m, K = {kb..kb+7, kb+16..kb+23}, kb = (lane>=16) ? 8 : 0.
// ---------------------------------------------------------------------------
__device__ inline v16h load_frag(const _Float16* __restrict__ p, int ldk) {
  const int lane = threadIdx.x & 31;
  const _Float16* rp = p + (lane & 15) * ldk + ((lane >> 4) << 3);
  v8h lo = *(const v8h*)rp;          // K = kb .. kb+7
  v8h hi = *(const v8h*)(rp + 16);   // K = kb+16 .. kb+23
  v16h r;
#pragma unroll
  for (int i = 0; i < 8; ++i) { r[i] = lo[i]; r[i + 8] = hi[i]; }
  return r;
}

__device__ inline v8f wmma_f16(v16h a, v16h b, v8f c) {
  return __builtin_amdgcn_wmma_f32_16x16x32_f16(
      false, a, false, b, (short)0, c, false, false);
}

__device__ inline unsigned lds_off_of(const void* p) {
  // Generic pointers to __shared__ carry the LDS byte offset in addr[31:0]
  // (aperture occupies addr[63:32], ISA 10.2 aperture check).
  return (unsigned)(unsigned long long)(uintptr_t)p;
}

// ---------------------------------------------------------------------------
// fp32 -> f16 elementwise convert
// ---------------------------------------------------------------------------
__global__ void cvt_f32_f16(const float* __restrict__ src,
                            _Float16* __restrict__ dst, int n) {
  int i = blockIdx.x * blockDim.x + threadIdx.x;
  int stride = gridDim.x * blockDim.x;
  for (; i < n; i += stride) dst[i] = (_Float16)src[i];
}

// ---------------------------------------------------------------------------
// GEMM: Y[M=4096, N=1024] = X(f16,[M,K]) @ Wt(f16,[N,K] row-major) + bias
//   mode 0: out16[((b*H+h)*S + s)*DK + d]   (Q / K head layout)
//   mode 1: out16[((b*H+h)*DK + d)*S + s]   (V transposed head layout)
//   mode 2: out32[m*D + n] fp32             (final output)
// Block: 256 threads = 8 waves. Tile 128(M) x 64(N), K-step 32.
// B tile DMA'd into double-buffered LDS by the TDM (wave 0 issues).
// ---------------------------------------------------------------------------
__global__ __launch_bounds__(256) void gemm_wmma(
    const _Float16* __restrict__ X, const _Float16* __restrict__ Wt,
    const float* __restrict__ bias, _Float16* __restrict__ out16,
    float* __restrict__ out32, int mode) {
  __shared__ _Float16 Bt[2][64][32];   // double-buffered [n][k] tile, 8 KB
  const int K    = D_;
  const int n0   = blockIdx.x * 64;
  const int m0   = blockIdx.y * 128;
  const int tid  = threadIdx.x;
  const int w    = tid >> 5;
  const int lane = tid & 31;

  const unsigned bt_off = lds_off_of(&Bt[0][0][0]);
  const _Float16* wrow  = Wt + (size_t)n0 * K;
  const _Float16* arow  = X + (size_t)(m0 + w * 16) * K;

  v8f acc[4] = {{}, {}, {}, {}};

  if (w == 0)
    tdm_load_2d(bt_off, wrow, /*tile*/ 32, 64, /*stride*/ K, K, D_);

  int cur = 0;
  for (int k0 = 0; k0 < K; k0 += 32) {
    if (w == 0) {
      if (k0 + 32 < K) {
        tdm_load_2d(bt_off + (unsigned)((cur ^ 1) * 4096), wrow + k0 + 32,
                    32, 64, K, K, D_);
        __builtin_amdgcn_s_wait_tensorcnt(1);  // in-order: current tile done
      } else {
        __builtin_amdgcn_s_wait_tensorcnt(0);
      }
    }
    __builtin_prefetch(arow + k0 + 64, 0, 1);  // global_prefetch_b8 (next A)
    __syncthreads();

    v16h af = load_frag(arow + k0, K);
#pragma unroll
    for (int a = 0; a < 4; ++a) {
      v16h bf = load_frag(&Bt[cur][a * 16][0], 32);
      acc[a] = wmma_f16(af, bf, acc[a]);
    }
    __syncthreads();   // all reads done before TDM overwrites this buffer
    cur ^= 1;
  }

  // epilogue: C layout -> row M = r + 8*(lane>=16), col N = lane%16
  const int hl  = lane >> 4;
  const int col = lane & 15;
#pragma unroll
  for (int a = 0; a < 4; ++a) {
    const int n    = n0 + a * 16 + col;
    const float bv = bias[n];
#pragma unroll
    for (int r = 0; r < 8; ++r) {
      const int m     = m0 + w * 16 + r + hl * 8;
      const float val = acc[a][r] + bv;
      if (mode == 2) {
        out32[(long)m * D_ + n] = val;
      } else {
        const int b = m >> 11, s = m & (S_ - 1);
        const int h = n >> 6,  d = n & 63;
        long idx;
        if (mode == 0) idx = (((long)(b * H_ + h) * S_ + s) * DK_ + d);
        else           idx = (((long)(b * H_ + h) * DK_ + d) * S_ + s);
        out16[idx] = (_Float16)val;
      }
    }
  }
}

// ---------------------------------------------------------------------------
// Flash attention (causal). One (b,h) x 128-query block per workgroup;
// wave w owns 16 query rows and the full DK=64 output accumulator.
// K and V^T tiles (64 keys) are TDM-DMA'd into double-buffered LDS and
// shared by all 8 waves; P restaged via per-wave LDS tile (C->A layout).
// ---------------------------------------------------------------------------
__global__ __launch_bounds__(256) void flash_attn(
    const _Float16* __restrict__ Q,   // [BH, S, DK]
    const _Float16* __restrict__ Kh,  // [BH, S, DK]
    const _Float16* __restrict__ VT,  // [BH, DK, S]
    _Float16* __restrict__ O16) {     // [B, S, D]
  __shared__ _Float16 Kt[2][64][64];  // keys x d, 16 KB
  __shared__ _Float16 Vt[2][64][64];  // d x keys, 16 KB
  __shared__ _Float16 Pt[8][16][64];  // per-wave P tile, 16 KB
  const int tid  = threadIdx.x;
  const int w    = tid >> 5;
  const int lane = tid & 31;
  const int hl   = lane >> 4;
  const int col  = lane & 15;
  const int qb   = blockIdx.x;              // 0..15
  const int bh   = blockIdx.y;              // 0..31
  const int q0   = qb * 128 + w * 16;       // wave's first query row
  const float scale = 0.125f;               // 1/sqrt(64)

  const unsigned kt_off = lds_off_of(&Kt[0][0][0]);
  const unsigned vt_off = lds_off_of(&Vt[0][0][0]);
  const _Float16* kg = Kh + (size_t)bh * S_ * DK_;  // [S, DK] rows
  const _Float16* vg = VT + (size_t)bh * DK_ * S_;  // [DK, S] rows

  const _Float16* qbase = Q + ((size_t)bh * S_ + q0) * DK_;
  v16h qf0 = load_frag(qbase, DK_);
  v16h qf1 = load_frag(qbase + 32, DK_);

  v8f o[4] = {{}, {}, {}, {}};
  float mrow[8], lrow[8];
#pragma unroll
  for (int r = 0; r < 8; ++r) { mrow[r] = -1e30f; lrow[r] = 0.f; }

  const int q_hi = q0 + 15;         // wave-uniform causal bound
  const int nkb  = qb * 2 + 2;      // block-uniform #key blocks (covers q0+127)

  if (w == 0) {
    tdm_load_2d(kt_off, kg, 64, 64, DK_, DK_, S_);          // K tile kb=0
    tdm_load_2d(vt_off, vg, 64, 64, S_,  S_,  DK_);         // V^T tile kb=0
  }

  int cur = 0;
  for (int kb = 0; kb < nkb; ++kb) {
    const int kstart = kb * 64;
    if (w == 0) {
      if (kb + 1 < nkb) {
        const unsigned nb = (unsigned)((cur ^ 1) * 8192);
        tdm_load_2d(kt_off + nb, kg + (size_t)(kstart + 64) * DK_,
                    64, 64, DK_, DK_, S_);
        tdm_load_2d(vt_off + nb, vg + kstart + 64,
                    64, 64, S_, S_, DK_);
        __builtin_amdgcn_s_wait_tensorcnt(2);  // 2 oldest (current) done
      } else {
        __builtin_amdgcn_s_wait_tensorcnt(0);
      }
    }
    __syncthreads();

    if (kstart <= q_hi) {   // wave-uniform causal skip (barriers still hit)
      // ---- scores: S(16x64) = Q(16x64) @ K^T ----
      v8f sc[4] = {{}, {}, {}, {}};
#pragma unroll
      for (int a = 0; a < 4; ++a) {
        sc[a] = wmma_f16(qf0, load_frag(&Kt[cur][a * 16][0], 64), sc[a]);
        sc[a] = wmma_f16(qf1, load_frag(&Kt[cur][a * 16][32], 64), sc[a]);
      }

      // ---- causal mask + online softmax ----
#pragma unroll
      for (int r = 0; r < 8; ++r) {
        const int qrow = q0 + r + hl * 8;
        float sv[4];
        float mx = -1e30f;
#pragma unroll
        for (int a = 0; a < 4; ++a) {
          const int key = kstart + a * 16 + col;
          float s = sc[a][r] * scale;
          if (key > qrow) s = -1e30f;
          sv[a] = s;
          mx = fmaxf(mx, s);
        }
        mx = fmaxf(mx, __shfl_xor(mx, 1, 32));
        mx = fmaxf(mx, __shfl_xor(mx, 2, 32));
        mx = fmaxf(mx, __shfl_xor(mx, 4, 32));
        mx = fmaxf(mx, __shfl_xor(mx, 8, 32));
        const float mnew  = fmaxf(mrow[r], mx);
        const float alpha = __expf(mrow[r] - mnew);
        mrow[r] = mnew;

        float rs = 0.f;
#pragma unroll
        for (int a = 0; a < 4; ++a) {
          const float p = __expf(sv[a] - mnew);
          rs += p;
          Pt[w][r + hl * 8][a * 16 + col] = (_Float16)p;
        }
        rs += __shfl_xor(rs, 1, 32);
        rs += __shfl_xor(rs, 2, 32);
        rs += __shfl_xor(rs, 4, 32);
        rs += __shfl_xor(rs, 8, 32);
        lrow[r] = lrow[r] * alpha + rs;
#pragma unroll
        for (int a = 0; a < 4; ++a) o[a][r] *= alpha;
      }

      // ---- O += P(16x64) @ V(64x64) via V^T rows in LDS ----
#pragma unroll
      for (int kk = 0; kk < 2; ++kk) {
        v16h pf = load_frag(&Pt[w][0][kk * 32], 64);
#pragma unroll
        for (int a = 0; a < 4; ++a) {
          o[a] = wmma_f16(pf, load_frag(&Vt[cur][a * 16][kk * 32], 64), o[a]);
        }
      }
    }
    __syncthreads();   // all reads done before TDM overwrites this buffer
    cur ^= 1;
  }

  // ---- normalize + scatter back to [B, S, D] ----
  float inv[8];
#pragma unroll
  for (int r = 0; r < 8; ++r) inv[r] = 1.f / lrow[r];
  const int b = bh >> 4;
  const int h = bh & 15;
#pragma unroll
  for (int a = 0; a < 4; ++a) {
    const int d = a * 16 + col;
#pragma unroll
    for (int r = 0; r < 8; ++r) {
      const int s = q0 + r + hl * 8;
      O16[((long)b * S_ + s) * D_ + h * DK_ + d] =
          (_Float16)(o[a][r] * inv[r]);
    }
  }
}

// ---------------------------------------------------------------------------
extern "C" void kernel_launch(void* const* d_in, const int* in_sizes, int n_in,
                              void* d_out, int out_size, void* d_ws,
                              size_t ws_size, hipStream_t stream) {
  const float* query  = (const float*)d_in[0];
  const float* key_in = (const float*)d_in[1];
  const float* value  = (const float*)d_in[2];
  const float* Wq = (const float*)d_in[3];
  const float* bq = (const float*)d_in[4];
  const float* Wk = (const float*)d_in[5];
  const float* bk = (const float*)d_in[6];
  const float* Wv = (const float*)d_in[7];
  const float* bv = (const float*)d_in[8];
  const float* Wo = (const float*)d_in[9];
  const float* bo = (const float*)d_in[10];
  float* out = (float*)d_out;

  const size_t MD = (size_t)MTOT * D_;  // 4M elems
  const size_t DD = (size_t)D_ * D_;    // 1M elems
  _Float16* ws   = (_Float16*)d_ws;
  _Float16* Xq   = ws;
  _Float16* Xk   = Xq + MD;
  _Float16* Xv   = Xk + MD;
  _Float16* Wq16 = Xv + MD;
  _Float16* Wk16 = Wq16 + DD;
  _Float16* Wv16 = Wk16 + DD;
  _Float16* Wo16 = Wv16 + DD;
  _Float16* Qh   = Wo16 + DD;
  _Float16* Kh   = Qh + MD;
  _Float16* VT   = Kh + MD;
  _Float16* AO   = VT + MD;

  cvt_f32_f16<<<4096, 256, 0, stream>>>(query,  Xq, (int)MD);
  cvt_f32_f16<<<4096, 256, 0, stream>>>(key_in, Xk, (int)MD);
  cvt_f32_f16<<<4096, 256, 0, stream>>>(value,  Xv, (int)MD);
  cvt_f32_f16<<<1024, 256, 0, stream>>>(Wq, Wq16, (int)DD);
  cvt_f32_f16<<<1024, 256, 0, stream>>>(Wk, Wk16, (int)DD);
  cvt_f32_f16<<<1024, 256, 0, stream>>>(Wv, Wv16, (int)DD);
  cvt_f32_f16<<<1024, 256, 0, stream>>>(Wo, Wo16, (int)DD);

  dim3 gg(D_ / 64, MTOT / 128);  // (16, 32)
  gemm_wmma<<<gg, 256, 0, stream>>>(Xq, Wq16, bq, Qh, nullptr, 0);
  gemm_wmma<<<gg, 256, 0, stream>>>(Xk, Wk16, bk, Kh, nullptr, 0);
  gemm_wmma<<<gg, 256, 0, stream>>>(Xv, Wv16, bv, VT, nullptr, 1);

  dim3 ga(S_ / 128, B_ * H_);  // (16, 32)
  flash_attn<<<ga, 256, 0, stream>>>(Qh, Kh, VT, AO);

  gemm_wmma<<<gg, 256, 0, stream>>>(AO, Wo16, bo, nullptr, out, 2);
}
